// Res21D_6828998001209
// MI455X (gfx1250) — compile-verified, matrix operand order checked by reading, not run on hardware
//
#include <hip/hip_runtime.h>
#include <hip/hip_bf16.h>

typedef __bf16 bf16;
typedef __attribute__((ext_vector_type(16))) __bf16 v16bf;
typedef __attribute__((ext_vector_type(8)))  __bf16 v8bf;
typedef __attribute__((ext_vector_type(8)))  float  v8f;

#define BB   32
#define SS   100
#define KWIN 5
#define EE   128
#define HH   225
#define HRR  900
#define BS   3200            // B*S
#define DHW  1125            // 5*15*15
#define EPSV 1e-5f

#define CDIV(a,b) (((a)+(b)-1)/(b))

__device__ __forceinline__ bf16 f2bf(float f) {
    union { float f; unsigned u; } x; x.f = f;
    unsigned r = (x.u + 0x7FFFu + ((x.u >> 16) & 1u)) >> 16;
    union { unsigned short s; bf16 b; } y; y.s = (unsigned short)r;
    return y.b;
}
__device__ __forceinline__ float sigm(float x) { return 1.f / (1.f + __expf(-x)); }

__device__ __forceinline__ v16bf frag_from(const bf16* p, int lg) {
    v8bf lo = *(const v8bf*)(p + 8 * lg);
    v8bf hi = *(const v8bf*)(p + 16 + 8 * lg);
    v16bf f;
    #pragma unroll
    for (int i = 0; i < 8; ++i) { f[i] = lo[i]; f[8 + i] = hi[i]; }
    return f;
}

// ---------------------------------------------------------------------------
// Generic bf16 WMMA GEMM:  Cout[M,N] = A[M,Kp] @ Bt[N,Kp]^T (+ bias) (+ Cin)
// Wave tile = (16*MT) x (16*NT). M multiple of 16*MT, Kp multiple of 32,
// Bt zero-padded to ceil(N/(16*NT))*(16*NT) rows.
// ---------------------------------------------------------------------------
template<int MT, int NT>
__global__ void gemm_wmma(const bf16* __restrict__ A, const bf16* __restrict__ Bt,
                          const float* __restrict__ Cin, const float* __restrict__ bias,
                          float* __restrict__ Cout,
                          int M, int N, int Kp, int ldcin, int ldcout)
{
    const int lane = threadIdx.x & 31;
    const int wave = (blockIdx.x * blockDim.x + threadIdx.x) >> 5;
    const int ntg = CDIV(N, 16 * NT);
    const int mtg = M / (16 * MT);
    if (wave >= mtg * ntg) return;
    const int m0 = (wave / ntg) * 16 * MT;
    const int n0 = (wave % ntg) * 16 * NT;
    const int lg = lane >> 4;
    const int r  = lane & 15;

    v8f z = {0.f,0.f,0.f,0.f,0.f,0.f,0.f,0.f};
    v8f acc[MT][NT];
    #pragma unroll
    for (int mt = 0; mt < MT; ++mt)
        #pragma unroll
        for (int nt = 0; nt < NT; ++nt) acc[mt][nt] = z;

    for (int kb = 0; kb < Kp; kb += 32) {
        v16bf af[MT];
        #pragma unroll
        for (int mt = 0; mt < MT; ++mt) {
            const bf16* arow = A + (size_t)(m0 + mt * 16 + r) * Kp + kb;
            if (kb + 32 < Kp) __builtin_prefetch(arow + 32, 0, 1);   // global_prefetch_b8
            af[mt] = frag_from(arow, lg);
        }
        #pragma unroll
        for (int nt = 0; nt < NT; ++nt) {
            const bf16* brow = Bt + (size_t)(n0 + nt * 16 + r) * Kp + kb;
            if (kb + 32 < Kp) __builtin_prefetch(brow + 32, 0, 1);
            v16bf bfv = frag_from(brow, lg);
            #pragma unroll
            for (int mt = 0; mt < MT; ++mt)
                acc[mt][nt] = __builtin_amdgcn_wmma_f32_16x16x32_bf16(
                    false, af[mt], false, bfv, (short)0, acc[mt][nt], false, false);
        }
    }

    #pragma unroll
    for (int mt = 0; mt < MT; ++mt)
        #pragma unroll
        for (int nt = 0; nt < NT; ++nt) {
            int n = n0 + nt * 16 + r;
            if (n >= N) continue;
            float bv = bias ? bias[n] : 0.f;
            #pragma unroll
            for (int v = 0; v < 8; ++v) {
                int m = m0 + mt * 16 + v + 8 * lg;
                float o = acc[mt][nt][v] + bv;
                if (Cin) o += Cin[(size_t)m * ldcin + n];
                Cout[(size_t)m * ldcout + n] = o;
            }
        }
}

// ---------------------------------------------------------------------------
// Big-GEMM variant: A panel (32 rows x 32 K) staged in LDS per 8-wave block
// via global_load_async_to_lds_b128 (double-buffered, ASYNCcnt + barrier).
// Block tile 32 x 256; wave tile 32 x 32 (MT=2,NT=2). M mult of 32,
// Bt padded to nbn*256 rows. EXEC all-ones everywhere (exact grid).
// ---------------------------------------------------------------------------
__global__ void gemm_wmma_ldsA(const bf16* __restrict__ A, const bf16* __restrict__ Bt,
                               const float* __restrict__ bias, float* __restrict__ Cout,
                               int N, int Kp, int ldcout, int nbn)
{
    __shared__ __align__(16) bf16 shA[2][32 * 32];
    const int tid  = threadIdx.x;
    const int lane = tid & 31;
    const int wv   = tid >> 5;
    const int m0b  = (blockIdx.x / nbn) * 32;
    const int n0   = (blockIdx.x % nbn) * 256 + wv * 32;
    const int lg = lane >> 4;
    const int r  = lane & 15;
    const int rr = tid >> 2;          // staging: row within 32-row panel
    const int ch = tid & 3;           // staging: 16B chunk within 64B row

    v8f z = {0.f,0.f,0.f,0.f,0.f,0.f,0.f,0.f};
    v8f acc[2][2] = {{z, z}, {z, z}};

    // stage first panel
    if (tid < 128) {
        const bf16* src = A + (size_t)(m0b + rr) * Kp + ch * 8;
        unsigned dst = (unsigned)(size_t)(&shA[0][0]) + rr * 64 + ch * 16;
        asm volatile("global_load_async_to_lds_b128 %0, %1, off"
                     :: "v"(dst), "v"((unsigned long long)(size_t)src) : "memory");
    }
    asm volatile("s_wait_asynccnt 0x0" ::: "memory");
    __syncthreads();

    int cur = 0;
    for (int kb = 0; kb < Kp; kb += 32, cur ^= 1) {
        if (kb + 32 < Kp && tid < 128) {          // stage next panel async
            const bf16* src = A + (size_t)(m0b + rr) * Kp + (kb + 32) + ch * 8;
            unsigned dst = (unsigned)(size_t)(&shA[cur ^ 1][0]) + rr * 64 + ch * 16;
            asm volatile("global_load_async_to_lds_b128 %0, %1, off"
                         :: "v"(dst), "v"((unsigned long long)(size_t)src) : "memory");
        }
        v16bf af[2];
        #pragma unroll
        for (int mt = 0; mt < 2; ++mt)            // A frags from LDS (ds_load_b128)
            af[mt] = frag_from(&shA[cur][(mt * 16 + r) * 32], lg);
        #pragma unroll
        for (int nt = 0; nt < 2; ++nt) {
            const bf16* brow = Bt + (size_t)(n0 + nt * 16 + r) * Kp + kb;
            if (kb + 32 < Kp) __builtin_prefetch(brow + 32, 0, 1);
            v16bf bfv = frag_from(brow, lg);
            #pragma unroll
            for (int mt = 0; mt < 2; ++mt)
                acc[mt][nt] = __builtin_amdgcn_wmma_f32_16x16x32_bf16(
                    false, af[mt], false, bfv, (short)0, acc[mt][nt], false, false);
        }
        asm volatile("s_wait_asynccnt 0x0" ::: "memory");
        __syncthreads();
    }

    #pragma unroll
    for (int mt = 0; mt < 2; ++mt)
        #pragma unroll
        for (int nt = 0; nt < 2; ++nt) {
            int n = n0 + nt * 16 + r;
            if (n >= N) continue;
            float bv = bias ? bias[n] : 0.f;
            #pragma unroll
            for (int v = 0; v < 8; ++v)
                Cout[(size_t)(m0b + mt * 16 + v + 8 * lg) * ldcout + n] = acc[mt][nt][v] + bv;
        }
}

// f32 [N,Kc] -> bf16 zero-padded [Np,Kp]
__global__ void pack_bf16(const float* __restrict__ W, bf16* __restrict__ P,
                          int N, int Kc, int Np, int Kp)
{
    int i = blockIdx.x * blockDim.x + threadIdx.x;
    if (i >= Np * Kp) return;
    int n = i / Kp, k = i % Kp;
    float v = (n < N && k < Kc) ? W[(size_t)n * Kc + k] : 0.f;
    P[i] = f2bf(v);
}

__global__ void vadd_f32(const float* a, const float* b, float* o, int n)
{
    int i = blockIdx.x * blockDim.x + threadIdx.x;
    if (i < n) o[i] = a[i] + b[i];
}
__global__ void copy_f32(const float* a, float* o, int n)
{
    int i = blockIdx.x * blockDim.x + threadIdx.x;
    if (i < n) o[i] = a[i];
}

__global__ void embed_gather(const int* __restrict__ x, const float* __restrict__ embed,
                             bf16* __restrict__ out)
{
    int i = blockIdx.x * blockDim.x + threadIdx.x;
    if (i >= BS * EE) return;
    int r = i / EE, e = i % EE;
    out[i] = f2bf(embed[(size_t)x[r] * EE + e]);
}

// LSTM pointwise (gate order i,f,g,o)
__global__ void lstm_point(const float* __restrict__ zstep, float* __restrict__ c,
                           bf16* __restrict__ hprev_pad, int Hc, int Hpad,
                           bf16* __restrict__ hseq_pad, float* __restrict__ hseq_f32, int t)
{
    int i = blockIdx.x * blockDim.x + threadIdx.x;
    if (i >= BB * Hc) return;
    int b = i / Hc, j = i % Hc;
    const float* zb = zstep + (size_t)b * 4 * Hc;
    float ig = sigm(zb[j]);
    float fg = sigm(zb[Hc + j]);
    float gg = tanhf(zb[2 * Hc + j]);
    float og = sigm(zb[3 * Hc + j]);
    float cn = fg * c[i] + ig * gg;
    float hn = og * tanhf(cn);
    c[i] = cn;
    hprev_pad[(size_t)b * Hpad + j] = f2bf(hn);
    size_t row = (size_t)b * SS + t;
    if (hseq_f32) hseq_f32[row * Hc + j] = hn;
    if (hseq_pad) hseq_pad[row * Hpad + j] = f2bf(hn);
}

// hs [B*S, 225] -> clips [B*S, 5*225], left zero padding in time
__global__ void build_clips(const float* __restrict__ hs, float* __restrict__ clips)
{
    int i = blockIdx.x * blockDim.x + threadIdx.x;
    if (i >= BS * DHW) return;
    int n = i / DHW, q = i % DHW;
    int d = q / HH, p = q % HH;
    int b = n / SS, t = n % SS;
    int fr = t - (KWIN - 1) + d;
    clips[i] = (fr >= 0) ? hs[((size_t)b * SS + fr) * HH + p] : 0.f;
}

// direct 3x3x3 conv, pad=1, over [BS, Cin, 5, 15, 15]
__global__ void conv3d_k3(const float* __restrict__ in, const float* __restrict__ w,
                          float* __restrict__ out, int Cin, int Cout)
{
    int i = blockIdx.x * blockDim.x + threadIdx.x;
    int total = BS * Cout * DHW;
    if (i >= total) return;
    int p  = i % DHW;
    int co = (i / DHW) % Cout;
    int n  = i / (DHW * Cout);
    int wc = p % 15, h = (p / 15) % 15, d = p / 225;
    float acc = 0.f;
    for (int ci = 0; ci < Cin; ++ci) {
        const float* inb = in + ((size_t)n * Cin + ci) * DHW;
        const float* wb  = w + ((size_t)co * Cin + ci) * 27;
        #pragma unroll
        for (int kd = 0; kd < 3; ++kd) {
            int dd = d + kd - 1; if (dd < 0 || dd >= 5) continue;
            #pragma unroll
            for (int kh = 0; kh < 3; ++kh) {
                int hh2 = h + kh - 1; if (hh2 < 0 || hh2 >= 15) continue;
                #pragma unroll
                for (int kw = 0; kw < 3; ++kw) {
                    int ww2 = wc + kw - 1; if (ww2 < 0 || ww2 >= 15) continue;
                    acc += inb[(dd * 15 + hh2) * 15 + ww2] * wb[(kd * 3 + kh) * 3 + kw];
                }
            }
        }
    }
    out[i] = acc;
}

__global__ void conv3d_k1(const float* __restrict__ in, const float* __restrict__ w,
                          float* __restrict__ out, int Cin, int Cout)
{
    int i = blockIdx.x * blockDim.x + threadIdx.x;
    int total = BS * Cout * DHW;
    if (i >= total) return;
    int p  = i % DHW;
    int co = (i / DHW) % Cout;
    int n  = i / (DHW * Cout);
    float acc = 0.f;
    for (int ci = 0; ci < Cin; ++ci)
        acc += in[((size_t)n * Cin + ci) * DHW + p] * w[co * Cin + ci];
    out[i] = acc;
}

// BN stats: per-channel sum/sumsq over [BS, Cc, DHW] via LDS tree + atomics
__global__ void bn_stats(const float* __restrict__ buf, float* __restrict__ acc, int Cc)
{
    long total = (long)BS * Cc * DHW;
    float s[4] = {0, 0, 0, 0}, q[4] = {0, 0, 0, 0};
    for (long i = (long)blockIdx.x * blockDim.x + threadIdx.x; i < total;
         i += (long)gridDim.x * blockDim.x) {
        int c = (int)((i / DHW) % Cc);
        float v = buf[i];
        s[c] += v; q[c] += v * v;
    }
    __shared__ float sh[512];
    for (int c = 0; c < Cc; ++c) {
        sh[threadIdx.x] = s[c]; sh[256 + threadIdx.x] = q[c];
        __syncthreads();
        for (int off = 128; off > 0; off >>= 1) {
            if ((int)threadIdx.x < off) {
                sh[threadIdx.x] += sh[threadIdx.x + off];
                sh[256 + threadIdx.x] += sh[256 + threadIdx.x + off];
            }
            __syncthreads();
        }
        if (threadIdx.x == 0) {
            atomicAdd(&acc[2 * c], sh[0]);
            atomicAdd(&acc[2 * c + 1], sh[256]);
        }
        __syncthreads();
    }
}

__global__ void bn_relu_inplace(float* __restrict__ buf, const float* __restrict__ acc,
                                const float* __restrict__ g, const float* __restrict__ b,
                                int Cc, float invcnt)
{
    long total = (long)BS * Cc * DHW;
    long i = (long)blockIdx.x * blockDim.x + threadIdx.x;
    if (i >= total) return;
    int c = (int)((i / DHW) % Cc);
    float m = acc[2 * c] * invcnt;
    float v = acc[2 * c + 1] * invcnt - m * m;
    float y = (buf[i] - m) * rsqrtf(v + EPSV) * g[c] + b[c];
    buf[i] = y > 0.f ? y : 0.f;
}

// out = relu( bn(x) + (dsb ? bn(dsb) : res) )
__global__ void bn_combine(const float* __restrict__ x, const float* __restrict__ accx,
                           const float* __restrict__ gx, const float* __restrict__ bx,
                           const float* __restrict__ res,
                           const float* __restrict__ dsb, const float* __restrict__ accd,
                           const float* __restrict__ gd, const float* __restrict__ bd,
                           float* __restrict__ out, int Cc, float invcnt)
{
    long total = (long)BS * Cc * DHW;
    long i = (long)blockIdx.x * blockDim.x + threadIdx.x;
    if (i >= total) return;
    int c = (int)((i / DHW) % Cc);
    float m = accx[2 * c] * invcnt;
    float v = accx[2 * c + 1] * invcnt - m * m;
    float y = (x[i] - m) * rsqrtf(v + EPSV) * gx[c] + bx[c];
    float r;
    if (dsb) {
        float md = accd[2 * c] * invcnt;
        float vd = accd[2 * c + 1] * invcnt - md * md;
        r = (dsb[i] - md) * rsqrtf(vd + EPSV) * gd[c] + bd[c];
    } else {
        r = res[i];
    }
    y += r;
    out[i] = y > 0.f ? y : 0.f;
}

// ---------------------------------------------------------------------------
template<int MT, int NT>
static inline void launch_gemm(const bf16* A, const bf16* Bt, const float* Cin,
                               const float* bias, float* Cout,
                               int M, int N, int Kp, int ldcin, int ldcout,
                               hipStream_t s)
{
    int waves = (M / (16 * MT)) * CDIV(N, 16 * NT);
    int blocks = CDIV(waves * 32, 256);
    gemm_wmma<MT, NT><<<blocks, 256, 0, s>>>(A, Bt, Cin, bias, Cout, M, N, Kp, ldcin, ldcout);
}

extern "C" void kernel_launch(void* const* d_in, const int* in_sizes, int n_in,
                              void* d_out, int out_size, void* d_ws, size_t ws_size,
                              hipStream_t stream)
{
    (void)in_sizes; (void)n_in; (void)out_size; (void)ws_size;
    const int*   x     = (const int*)  d_in[0];
    const float* embed = (const float*)d_in[1];
    const float* h0f   = (const float*)d_in[2];
    const float* c0f   = (const float*)d_in[3];
    const float* Wih_f = (const float*)d_in[4];
    const float* Whh_f = (const float*)d_in[5];
    const float* bih_f = (const float*)d_in[6];
    const float* bhh_f = (const float*)d_in[7];
    const float* w1a = (const float*)d_in[8],  *g1a = (const float*)d_in[9],  *b1a = (const float*)d_in[10];
    const float* w1b = (const float*)d_in[11], *g1b = (const float*)d_in[12], *b1b = (const float*)d_in[13];
    const float* wds1= (const float*)d_in[14], *gds1= (const float*)d_in[15], *bds1= (const float*)d_in[16];
    const float* w3a = (const float*)d_in[17], *g3a = (const float*)d_in[18], *b3a = (const float*)d_in[19];
    const float* w3b = (const float*)d_in[20], *g3b = (const float*)d_in[21], *b3b = (const float*)d_in[22];
    const float* w4a = (const float*)d_in[23], *g4a = (const float*)d_in[24], *b4a = (const float*)d_in[25];
    const float* w4b = (const float*)d_in[26], *g4b = (const float*)d_in[27], *b4b = (const float*)d_in[28];
    const float* wds4= (const float*)d_in[29], *gds4= (const float*)d_in[30], *bds4= (const float*)d_in[31];
    const float* h0r   = (const float*)d_in[32];
    const float* c0r   = (const float*)d_in[33];
    const float* Wih_r = (const float*)d_in[34];
    const float* Whh_r = (const float*)d_in[35];
    const float* bih_r = (const float*)d_in[36];
    const float* bhh_r = (const float*)d_in[37];
    const float* Wfc   = (const float*)d_in[38];
    const float* bfc   = (const float*)d_in[39];
    float* out = (float*)d_out;

    // ---- workspace layout ----
    char* w = (char*)d_ws;
    size_t off = 0;
    auto alloc = [&](size_t bytes) -> char* {
        char* p = w + off; off = (off + bytes + 255) & ~(size_t)255; return p;
    };
    bf16* wihf_p = (bf16*)alloc((size_t)960  * 128  * 2);
    bf16* whhf_p = (bf16*)alloc((size_t)960  * 256  * 2);
    bf16* wihr_p = (bf16*)alloc((size_t)3840 * 1152 * 2);   // padded to 15*256 rows
    bf16* whhr_p = (bf16*)alloc((size_t)3840 * 928  * 2);
    bf16* wfc_p  = (bf16*)alloc((size_t)128  * 928  * 2);
    float* bsum_f = (float*)alloc(900 * 4);
    float* bsum_r = (float*)alloc(3600 * 4);
    bf16* emb_bf  = (bf16*)alloc((size_t)BS * EE * 2);
    float* hs     = (float*)alloc((size_t)BS * HH * 4);
    bf16* hprevf  = (bf16*)alloc((size_t)BB * 256 * 2);
    float* cf     = (float*)alloc((size_t)BB * HH * 4);
    float* zstepf = (float*)alloc((size_t)BB * 900 * 4);
    float* clips  = (float*)alloc((size_t)BS * DHW * 4);    // reused as ds4 buf
    float* stats  = (float*)alloc(8 * 8 * 4);               // 8 sets of (sum,sumsq)x4ch
    bf16* feat_bf = (bf16*)alloc((size_t)BS * 1152 * 2);
    bf16* hprevr  = (bf16*)alloc((size_t)BB * 928 * 2);
    float* cr     = (float*)alloc((size_t)BB * HRR * 4);
    float* zstepr = (float*)alloc((size_t)BB * 3600 * 4);
    float* Abuf   = (float*)alloc((size_t)BS * 4 * DHW * 4);
    float* Bbuf   = (float*)alloc((size_t)BS * 4 * DHW * 4);
    float* Dbuf   = (float*)alloc((size_t)BS * 4 * DHW * 4);
    // aliases (lifetime-disjoint)
    float* Zf    = Dbuf;                 // [BS, 900]   front input-gates
    float* Zr    = Abuf;                 // [BS, 3600]  rear input-gates
    bf16*  hrall = (bf16*)Bbuf;          // [BS, 928]   rear h (padded bf16)
    float* ds4b  = clips;                // [BS, 1125]  block4 downsample out

    const float invcnt = 1.f / ((float)BS * DHW);
    auto statset = [&](int s) { return stats + 8 * s; };

    hipMemsetAsync(stats, 0, 8 * 8 * 4, stream);

    // ---- weight / bias prep ----
    pack_bf16<<<CDIV(960 * 128, 256), 256, 0, stream>>>(Wih_f, wihf_p, 900, 128, 960, 128);
    pack_bf16<<<CDIV(960 * 256, 256), 256, 0, stream>>>(Whh_f, whhf_p, 900, 225, 960, 256);
    pack_bf16<<<CDIV(3840 * 1152, 256), 256, 0, stream>>>(Wih_r, wihr_p, 3600, 1125, 3840, 1152);
    pack_bf16<<<CDIV(3840 * 928, 256), 256, 0, stream>>>(Whh_r, whhr_p, 3600, 900, 3840, 928);
    pack_bf16<<<CDIV(128 * 928, 256), 256, 0, stream>>>(Wfc, wfc_p, 110, 900, 128, 928);
    vadd_f32<<<CDIV(900, 256), 256, 0, stream>>>(bih_f, bhh_f, bsum_f, 900);
    vadd_f32<<<CDIV(3600, 256), 256, 0, stream>>>(bih_r, bhh_r, bsum_r, 3600);

    // ---- embedding + front input-gate GEMM: Zf = emb @ Wih_f^T + (bih+bhh) ----
    embed_gather<<<CDIV(BS * EE, 256), 256, 0, stream>>>(x, embed, emb_bf);
    launch_gemm<2, 4>(emb_bf, wihf_p, nullptr, bsum_f, Zf, BS, 900, 128, 0, 900, stream);

    // ---- front LSTM (sequential) ----
    pack_bf16<<<CDIV(BB * 256, 256), 256, 0, stream>>>(h0f, hprevf, BB, HH, BB, 256);
    copy_f32<<<CDIV(BB * HH, 256), 256, 0, stream>>>(c0f, cf, BB * HH);
    for (int t = 0; t < SS; ++t) {
        launch_gemm<1, 2>(hprevf, whhf_p, Zf + (size_t)t * 900, nullptr, zstepf,
                          BB, 900, 256, SS * 900, 900, stream);
        lstm_point<<<CDIV(BB * HH, 256), 256, 0, stream>>>(zstepf, cf, hprevf, HH, 256,
                                                           nullptr, hs, t);
    }

    // ---- clips + conv blocks ----
    build_clips<<<CDIV(BS * DHW, 256), 256, 0, stream>>>(hs, clips);

    // block1: 1 -> 4 (with downsample)
    conv3d_k3<<<CDIV(BS * 4 * DHW, 256), 256, 0, stream>>>(clips, w1a, Abuf, 1, 4);
    bn_stats<<<512, 256, 0, stream>>>(Abuf, statset(0), 4);
    bn_relu_inplace<<<CDIV(BS * 4 * DHW, 256), 256, 0, stream>>>(Abuf, statset(0), g1a, b1a, 4, invcnt);
    conv3d_k3<<<CDIV(BS * 4 * DHW, 256), 256, 0, stream>>>(Abuf, w1b, Bbuf, 4, 4);
    bn_stats<<<512, 256, 0, stream>>>(Bbuf, statset(1), 4);
    conv3d_k1<<<CDIV(BS * 4 * DHW, 256), 256, 0, stream>>>(clips, wds1, Dbuf, 1, 4);
    bn_stats<<<512, 256, 0, stream>>>(Dbuf, statset(2), 4);
    bn_combine<<<CDIV(BS * 4 * DHW, 256), 256, 0, stream>>>(
        Bbuf, statset(1), g1b, b1b, nullptr, Dbuf, statset(2), gds1, bds1, Abuf, 4, invcnt);

    // block3: 4 -> 4 (identity residual)
    conv3d_k3<<<CDIV(BS * 4 * DHW, 256), 256, 0, stream>>>(Abuf, w3a, Bbuf, 4, 4);
    bn_stats<<<512, 256, 0, stream>>>(Bbuf, statset(3), 4);
    bn_relu_inplace<<<CDIV(BS * 4 * DHW, 256), 256, 0, stream>>>(Bbuf, statset(3), g3a, b3a, 4, invcnt);
    conv3d_k3<<<CDIV(BS * 4 * DHW, 256), 256, 0, stream>>>(Bbuf, w3b, Dbuf, 4, 4);
    bn_stats<<<512, 256, 0, stream>>>(Dbuf, statset(4), 4);
    bn_combine<<<CDIV(BS * 4 * DHW, 256), 256, 0, stream>>>(
        Dbuf, statset(4), g3b, b3b, Abuf, nullptr, nullptr, nullptr, nullptr, Abuf, 4, invcnt);

    // block4: 4 -> 1 (with downsample)
    conv3d_k3<<<CDIV(BS * 1 * DHW, 256), 256, 0, stream>>>(Abuf, w4a, Bbuf, 4, 1);
    bn_stats<<<512, 256, 0, stream>>>(Bbuf, statset(5), 1);
    bn_relu_inplace<<<CDIV(BS * 1 * DHW, 256), 256, 0, stream>>>(Bbuf, statset(5), g4a, b4a, 1, invcnt);
    conv3d_k3<<<CDIV(BS * 1 * DHW, 256), 256, 0, stream>>>(Bbuf, w4b, Dbuf, 1, 1);
    bn_stats<<<512, 256, 0, stream>>>(Dbuf, statset(6), 1);
    conv3d_k1<<<CDIV(BS * 1 * DHW, 256), 256, 0, stream>>>(Abuf, wds4, ds4b, 4, 1);
    bn_stats<<<512, 256, 0, stream>>>(ds4b, statset(7), 1);
    bn_combine<<<CDIV(BS * 1 * DHW, 256), 256, 0, stream>>>(
        Dbuf, statset(6), g4b, b4b, nullptr, ds4b, statset(7), gds4, bds4, Bbuf, 1, invcnt);

    // feat -> bf16 padded [BS, 1152]
    pack_bf16<<<CDIV(BS * 1152, 256), 256, 0, stream>>>(Bbuf, feat_bf, BS, DHW, BS, 1152);

    // ---- rear input-gate GEMM (dominant, async-LDS A staging) ----
    gemm_wmma_ldsA<<<(BS / 32) * 15, 256, 0, stream>>>(feat_bf, wihr_p, bsum_r, Zr,
                                                       3600, 1152, 3600, 15);

    // ---- rear LSTM (sequential) ----
    pack_bf16<<<CDIV(BB * 928, 256), 256, 0, stream>>>(h0r, hprevr, BB, HRR, BB, 928);
    copy_f32<<<CDIV(BB * HRR, 256), 256, 0, stream>>>(c0r, cr, BB * HRR);
    hipMemsetAsync(hrall, 0, (size_t)BS * 928 * 2, stream);   // zero K-pad columns
    for (int t = 0; t < SS; ++t) {
        launch_gemm<1, 2>(hprevr, whhr_p, Zr + (size_t)t * 3600, nullptr, zstepr,
                          BB, 3600, 928, SS * 3600, 3600, stream);
        lstm_point<<<CDIV(BB * HRR, 256), 256, 0, stream>>>(zstepr, cr, hprevr, HRR, 928,
                                                            hrall, nullptr, t);
    }

    // ---- final FC: out[3200,110] = hr @ Wfc^T + bfc ----
    launch_gemm<2, 4>(hrall, wfc_p, nullptr, bfc, out, BS, 110, 928, 0, 110, stream);
}
